// GenerativeDecoder_76192719831570
// MI455X (gfx1250) — matrix operand order repaired and checked by reading, not compiled, not run on hardware
//
#include <hip/hip_runtime.h>

// ---------------- model dims (compile-time constants) ----------------
#define B_SZ   16
#define T_LEN  200
#define S_LEN  100
#define E_DIM  256
#define H_DIM  8
#define DH     32
#define L_NUM  2
#define F_DIM  1024
#define V_SZ   50000
#define DT_DIM 768
#define DI_DIM 512
#define GH_DIM 64

typedef __attribute__((ext_vector_type(16))) __bf16 v16bf;
typedef __attribute__((ext_vector_type(8)))  __bf16 v8bf;
typedef __attribute__((ext_vector_type(8)))  float  v8f;

__device__ __forceinline__ __bf16 f2bf(float f) {
  unsigned u = __builtin_bit_cast(unsigned, f);
  u += 0x7FFFu + ((u >> 16) & 1u);            // round-to-nearest-even
  unsigned short h = (unsigned short)(u >> 16);
  return __builtin_bit_cast(__bf16, h);
}

// ---------------------------------------------------------------------
// GEMM: C[M,N] = act(A[M,K](bf16) @ Bt[N,K](bf16)^T + bias), WMMA bf16.
// Bt is the pre-transposed weight: row n holds output-column n (K contig),
// which gives this code's B-fragment order (lanes 0-15: K 0-15, lanes 16-31:
// K 16-31). A fragment: same order, lanes 0-15 K 0-15, lanes 16-31 K 16-31.
// 8 waves/block, wave = one 16x16 tile, block tile 32(M) x 64(N).
// K must be a multiple of 64 (true for all GEMMs here: 256/512/768/1024).
// ---------------------------------------------------------------------
__global__ void gemm_bf16(const __bf16* __restrict__ A, int lda,
                          const __bf16* __restrict__ Bt, int ldb,
                          const float* __restrict__ bias,
                          float* __restrict__ C, int ldc,
                          int M, int N, int K, int relu) {
  const int lane = threadIdx.x & 31;
  const int wave = threadIdx.x >> 5;
  const int tM = blockIdx.y * 32 + (wave >> 2) * 16;
  const int tN = blockIdx.x * 64 + (wave & 3) * 16;
  if (tM >= M || tN >= N) return;                 // wave-uniform guard (EXEC all-ones)

  const int mrow = tM + (lane & 15);
  const int ncol = tN + (lane & 15);
  const int kbA  = (lane >> 4) * 16;
  const int kbB  = (lane >> 4) * 16;
  const __bf16* arow = A  + (size_t)mrow * lda;
  const __bf16* brow = Bt + (size_t)ncol * ldb;

  v8f acc = {};
  for (int k = 0; k < K; k += 64) {
    if (k + 64 < K) {                             // prefetch next K slice
      __builtin_prefetch(arow + k + 64, 0, 0);
      __builtin_prefetch(brow + k + 64, 0, 0);
    }
#pragma unroll
    for (int kk = 0; kk < 64; kk += 32) {
      const v8bf a0 = *(const v8bf*)(arow + k + kk + kbA);
      const v8bf a1 = *(const v8bf*)(arow + k + kk + kbA + 8);
      const v16bf a = __builtin_shufflevector(a0, a1,
          0, 1, 2, 3, 4, 5, 6, 7, 8, 9, 10, 11, 12, 13, 14, 15);
      const v16bf b = *(const v16bf*)(brow + k + kk + kbB);
      acc = __builtin_amdgcn_wmma_f32_16x16x32_bf16(false, a, false, b,
                                                    (short)0, acc, false, false);
    }
  }
  const int rbase = tM + ((lane >> 4) ? 8 : 0);
  const float bv = bias ? bias[ncol] : 0.0f;
#pragma unroll
  for (int i = 0; i < 8; ++i) {
    float v = acc[i] + bv;
    if (relu) v = fmaxf(v, 0.0f);
    C[(size_t)(rbase + i) * ldc + ncol] = v;
  }
}

// fp32 -> bf16 elementwise
__global__ void cvt_kernel(const float* __restrict__ src, __bf16* __restrict__ dst, int n) {
  const int idx = blockIdx.x * blockDim.x + threadIdx.x;
  if (idx < n) dst[idx] = f2bf(src[idx]);
}

// transpose-convert weight: Wt[n*K + k] = bf16(W[k*ldw + n])
__global__ void cvtT_kernel(const float* __restrict__ W, int ldw,
                            __bf16* __restrict__ Wt, int K, int N) {
  const int idx = blockIdx.x * blockDim.x + threadIdx.x;
  if (idx >= N * K) return;
  const int n = idx / K, k = idx - n * K;
  Wt[idx] = f2bf(W[(size_t)k * ldw + n]);
}

// gather rows by ids and convert: dst[r*lda + j] = bf16(src[ids[r]*lda + j])
__global__ void gather_cvt_kernel(const float* __restrict__ src, const int* __restrict__ ids,
                                  int lda, __bf16* __restrict__ dst, int rows) {
  const int idx = blockIdx.x * blockDim.x + threadIdx.x;
  if (idx >= rows * lda) return;
  const int r = idx / lda, j = idx - r * lda;
  dst[idx] = f2bf(src[(size_t)ids[r] * lda + j]);
}

// ---------------------------------------------------------------------
// Attention: one workgroup per (b,h). K/V head-slices staged in LDS,
// one thread per query row, single-pass online softmax.
// mode: 0 = none, 1 = causal (skip s>t), 2 = key padding (+ -1e9)
// ---------------------------------------------------------------------
__global__ void attn_kernel(const float* __restrict__ Q, int q_ld, int q_off,
                            const float* __restrict__ KV, int kv_ld, int k_off, int v_off,
                            const unsigned char* __restrict__ pad, int mode,
                            float* __restrict__ ctx, int Tq, int Tk) {
  __shared__ float sK[T_LEN * DH];
  __shared__ float sV[T_LEN * DH];
  const int h = blockIdx.x, b = blockIdx.y, tid = threadIdx.x;
  const int kvbase = b * Tk;
  for (int i = tid; i < Tk * DH; i += blockDim.x) {
    int s = i >> 5, d = i & 31;
    sK[i] = KV[(size_t)(kvbase + s) * kv_ld + k_off + h * DH + d];
    sV[i] = KV[(size_t)(kvbase + s) * kv_ld + v_off + h * DH + d];
  }
  __syncthreads();
  const int t = tid;
  if (t >= Tq) return;

  float q[DH];
  const float* qr = Q + (size_t)(b * Tq + t) * q_ld + q_off + h * DH;
#pragma unroll
  for (int d = 0; d < DH; ++d) q[d] = qr[d];

  float m = -3.0e38f, l = 0.0f, acc[DH];
#pragma unroll
  for (int d = 0; d < DH; ++d) acc[d] = 0.0f;

  const int send = (mode == 1) ? (t + 1) : Tk;
  for (int s = 0; s < send; ++s) {
    float lg = 0.0f;
#pragma unroll
    for (int d = 0; d < DH; ++d) lg += q[d] * sK[s * DH + d];
    lg *= 0.17677669529663687f;                   // 1/sqrt(32)
    if (mode == 2 && pad[b * Tk + s]) lg += -1.0e9f;
    float mn   = fmaxf(m, lg);
    float corr = __expf(m - mn);
    float p    = __expf(lg - mn);
    l = l * corr + p;
#pragma unroll
    for (int d = 0; d < DH; ++d) acc[d] = acc[d] * corr + p * sV[s * DH + d];
    m = mn;
  }
  const float inv = 1.0f / l;
  float* orow = ctx + (size_t)(b * Tq + t) * E_DIM + h * DH;
#pragma unroll
  for (int d = 0; d < DH; ++d) orow[d] = acc[d] * inv;
}

// out = LayerNorm(x + h) * s + b   (256 threads = one row)
__global__ void add_ln_kernel(const float* __restrict__ x, const float* __restrict__ h,
                              const float* __restrict__ s, const float* __restrict__ b,
                              float* __restrict__ out) {
  __shared__ float red[E_DIM];
  const int row = blockIdx.x, tid = threadIdx.x;
  const size_t idx = (size_t)row * E_DIM + tid;
  float v = x[idx] + h[idx];
  red[tid] = v; __syncthreads();
  for (int o = E_DIM / 2; o > 0; o >>= 1) { if (tid < o) red[tid] += red[tid + o]; __syncthreads(); }
  const float mu = red[0] * (1.0f / E_DIM); __syncthreads();
  const float d = v - mu;
  red[tid] = d * d; __syncthreads();
  for (int o = E_DIM / 2; o > 0; o >>= 1) { if (tid < o) red[tid] += red[tid + o]; __syncthreads(); }
  const float var = red[0] * (1.0f / E_DIM);
  out[idx] = d * rsqrtf(var + 1e-5f) * s[tid] + b[tid];
}

// x = item_emb[ids]*sqrt(E) + pos_emb
__global__ void embed_kernel(const int* __restrict__ ids, const float* __restrict__ item_emb,
                             const float* __restrict__ pos_emb, float* __restrict__ x) {
  const int idx = blockIdx.x * blockDim.x + threadIdx.x;
  if (idx >= B_SZ * T_LEN * E_DIM) return;
  const int e = idx & (E_DIM - 1);
  const int r = idx >> 8;
  const int t = r % T_LEN;
  x[idx] = item_emb[(size_t)ids[r] * E_DIM + e] * 16.0f + pos_emb[t * E_DIM + e];
}

// gate = softmax(relu(x@w1+b1)@w2+b2)  (64 threads = one row)
__global__ void gate_kernel(const float* __restrict__ x,
                            const float* __restrict__ w1, const float* __restrict__ b1,
                            const float* __restrict__ w2, const float* __restrict__ b2,
                            float* __restrict__ gate) {
  __shared__ float sX[E_DIM];
  __shared__ float sH[GH_DIM];
  __shared__ float sL[3];
  const int row = blockIdx.x, tid = threadIdx.x;
  for (int i = tid; i < E_DIM; i += GH_DIM) sX[i] = x[(size_t)row * E_DIM + i];
  __syncthreads();
  float a = b1[tid];
  for (int k = 0; k < E_DIM; ++k) a += sX[k] * w1[k * GH_DIM + tid];
  sH[tid] = fmaxf(a, 0.0f);
  __syncthreads();
  if (tid < 3) {
    float lg = b2[tid];
    for (int j = 0; j < GH_DIM; ++j) lg += sH[j] * w2[j * 3 + tid];
    sL[tid] = lg;
  }
  __syncthreads();
  if (tid == 0) {
    float mx = fmaxf(sL[0], fmaxf(sL[1], sL[2]));
    float e0 = __expf(sL[0] - mx), e1 = __expf(sL[1] - mx), e2 = __expf(sL[2] - mx);
    float inv = 1.0f / (e0 + e1 + e2);
    gate[row * 3 + 0] = e0 * inv;
    gate[row * 3 + 1] = e1 * inv;
    gate[row * 3 + 2] = e2 * inv;
  }
}

// hfin = g0*e_beh + g1*e_txt + g2*e_img
__global__ void mix_kernel(const float* __restrict__ g, const float* __restrict__ a,
                           const float* __restrict__ b, const float* __restrict__ c,
                           float* __restrict__ o) {
  const int idx = blockIdx.x * blockDim.x + threadIdx.x;
  if (idx >= B_SZ * T_LEN * E_DIM) return;
  const int r = idx >> 8;
  o[idx] = g[r * 3] * a[idx] + g[r * 3 + 1] * b[idx] + g[r * 3 + 2] * c[idx];
}

// ---------------------------------------------------------------------
extern "C" void kernel_launch(void* const* d_in, const int* in_sizes, int n_in,
                              void* d_out, int out_size, void* d_ws, size_t ws_size,
                              hipStream_t stream) {
  (void)in_sizes; (void)n_in; (void)out_size; (void)ws_size;
  // --- inputs (setup_inputs dict order) ---
  const int*           ids   = (const int*)d_in[0];
  const float*         enc   = (const float*)d_in[1];
  const unsigned char* pmask = (const unsigned char*)d_in[2];
  const float* item_emb = (const float*)d_in[3];
  const float* pos_emb  = (const float*)d_in[4];
  const float* sa_qkv_w = (const float*)d_in[5];
  const float* sa_qkv_b = (const float*)d_in[6];
  const float* sa_out_w = (const float*)d_in[7];
  const float* sa_out_b = (const float*)d_in[8];
  const float* ca_qkv_w = (const float*)d_in[9];
  const float* ca_qkv_b = (const float*)d_in[10];
  const float* ca_out_w = (const float*)d_in[11];
  const float* ca_out_b = (const float*)d_in[12];
  const float* n1_s = (const float*)d_in[13];
  const float* n1_b = (const float*)d_in[14];
  const float* n2_s = (const float*)d_in[15];
  const float* n2_b = (const float*)d_in[16];
  const float* n3_s = (const float*)d_in[17];
  const float* n3_b = (const float*)d_in[18];
  const float* ffn_w1 = (const float*)d_in[19];
  const float* ffn_b1 = (const float*)d_in[20];
  const float* ffn_w2 = (const float*)d_in[21];
  const float* ffn_b2 = (const float*)d_in[22];
  const float* g_w1 = (const float*)d_in[23];
  const float* g_b1 = (const float*)d_in[24];
  const float* g_w2 = (const float*)d_in[25];
  const float* g_b2 = (const float*)d_in[26];
  const float* be_w = (const float*)d_in[27];
  const float* be_b = (const float*)d_in[28];
  const float* text_mat = (const float*)d_in[29];
  const float* tp_w = (const float*)d_in[30];
  const float* tp_b = (const float*)d_in[31];
  const float* ct_qkv_w = (const float*)d_in[32];
  const float* ct_qkv_b = (const float*)d_in[33];
  const float* ct_out_w = (const float*)d_in[34];
  const float* ct_out_b = (const float*)d_in[35];
  const float* ce_w1 = (const float*)d_in[36];
  const float* ce_b1 = (const float*)d_in[37];
  const float* ce_w2 = (const float*)d_in[38];
  const float* ce_b2 = (const float*)d_in[39];
  const float* img_mat = (const float*)d_in[40];
  const float* ip_w = (const float*)d_in[41];
  const float* ip_b = (const float*)d_in[42];
  const float* im_qkv_w = (const float*)d_in[43];
  const float* im_qkv_b = (const float*)d_in[44];
  const float* im_out_w = (const float*)d_in[45];
  const float* im_out_b = (const float*)d_in[46];
  const float* ie_w1 = (const float*)d_in[47];
  const float* ie_b1 = (const float*)d_in[48];
  const float* ie_w2 = (const float*)d_in[49];
  const float* ie_b2 = (const float*)d_in[50];
  const float* fin_w = (const float*)d_in[51];
  const float* fin_b = (const float*)d_in[52];
  float* out = (float*)d_out;

  const int R  = B_SZ * T_LEN;                   // 3200 decoder rows
  const int RS = B_SZ * S_LEN;                   // 1600 encoder rows

  // --- fp32 workspace ---
  float* p = (float*)d_ws;
  auto take = [&](size_t n) { float* r = p; p += n; return r; };
  float* x    = take((size_t)R * E_DIM);
  float* qkv  = take((size_t)R * 3 * E_DIM);
  float* kvb  = take((size_t)RS * 2 * E_DIM);
  float* ctx  = take((size_t)R * E_DIM);
  float* proj = take((size_t)R * E_DIM);
  float* ffnh = take((size_t)R * F_DIM);
  float* th   = take((size_t)R * E_DIM);
  float* tkv  = take((size_t)R * 2 * E_DIM);
  float* gate = take((size_t)R * 4);
  float* ebeh = take((size_t)R * E_DIM);
  float* etxt = take((size_t)R * E_DIM);
  float* eimg = take((size_t)R * E_DIM);
  float* ehid = take((size_t)R * 2 * E_DIM);
  float* hfin = take((size_t)R * E_DIM);

  // --- bf16 workspace (activations + transposed weights) ---
  __bf16* q = (__bf16*)p;
  auto takeb = [&](size_t n) { __bf16* r = q; q += n; return r; };
  __bf16* x_bf    = takeb((size_t)R * E_DIM);
  __bf16* enc_bf  = takeb((size_t)RS * E_DIM);
  __bf16* ctx_bf  = takeb((size_t)R * E_DIM);
  __bf16* proj_bf = takeb((size_t)R * E_DIM);
  __bf16* ffnh_bf = takeb((size_t)R * F_DIM);
  __bf16* th_bf   = takeb((size_t)R * E_DIM);
  __bf16* gath_bf = takeb((size_t)R * DT_DIM);   // gathered text/img rows (DT >= DI)
  __bf16* ehid_bf = takeb((size_t)R * 2 * E_DIM);
  __bf16* hfin_bf = takeb((size_t)R * E_DIM);
  __bf16* saqkv_t[L_NUM], *caqkv_t[L_NUM], *saout_t[L_NUM], *caout_t[L_NUM];
  __bf16* ffn1_t[L_NUM], *ffn2_t[L_NUM];
  for (int l = 0; l < L_NUM; ++l) {
    saqkv_t[l] = takeb((size_t)3 * E_DIM * E_DIM);
    caqkv_t[l] = takeb((size_t)3 * E_DIM * E_DIM);
    saout_t[l] = takeb((size_t)E_DIM * E_DIM);
    caout_t[l] = takeb((size_t)E_DIM * E_DIM);
    ffn1_t[l]  = takeb((size_t)F_DIM * E_DIM);
    ffn2_t[l]  = takeb((size_t)E_DIM * F_DIM);
  }
  __bf16* be_t    = takeb((size_t)E_DIM * E_DIM);
  __bf16* tp_t    = takeb((size_t)E_DIM * DT_DIM);
  __bf16* ctqkv_t = takeb((size_t)3 * E_DIM * E_DIM);
  __bf16* ctout_t = takeb((size_t)E_DIM * E_DIM);
  __bf16* ce1_t   = takeb((size_t)2 * E_DIM * E_DIM);
  __bf16* ce2_t   = takeb((size_t)E_DIM * 2 * E_DIM);
  __bf16* ip_t    = takeb((size_t)E_DIM * DI_DIM);
  __bf16* imqkv_t = takeb((size_t)3 * E_DIM * E_DIM);
  __bf16* imout_t = takeb((size_t)E_DIM * E_DIM);
  __bf16* ie1_t   = takeb((size_t)2 * E_DIM * E_DIM);
  __bf16* ie2_t   = takeb((size_t)E_DIM * 2 * E_DIM);
  __bf16* fin_t   = takeb((size_t)V_SZ * E_DIM);

  auto gemm = [&](const __bf16* A, int lda, const __bf16* Bt, int ldb,
                  const float* bias, float* C, int ldc, int M, int N, int K, int relu) {
    dim3 grid((unsigned)((N + 63) / 64), (unsigned)((M + 31) / 32));
    gemm_bf16<<<grid, 256, 0, stream>>>(A, lda, Bt, ldb, bias, C, ldc, M, N, K, relu);
  };
  auto cvt = [&](const float* src, __bf16* dst, size_t n) {
    cvt_kernel<<<(unsigned)((n + 255) / 256), 256, 0, stream>>>(src, dst, (int)n);
  };
  auto cvtT = [&](const float* W, int ldw, __bf16* Wt, int K, int N) {
    size_t n = (size_t)K * N;
    cvtT_kernel<<<(unsigned)((n + 255) / 256), 256, 0, stream>>>(W, ldw, Wt, K, N);
  };
  const dim3 attn_grid(H_DIM, B_SZ);

  // --- one-time weight transpose-converts ---
  for (int l = 0; l < L_NUM; ++l) {
    cvtT(sa_qkv_w + (size_t)l * E_DIM * 3 * E_DIM, 3 * E_DIM, saqkv_t[l], E_DIM, 3 * E_DIM);
    cvtT(ca_qkv_w + (size_t)l * E_DIM * 3 * E_DIM, 3 * E_DIM, caqkv_t[l], E_DIM, 3 * E_DIM);
    cvtT(sa_out_w + (size_t)l * E_DIM * E_DIM, E_DIM, saout_t[l], E_DIM, E_DIM);
    cvtT(ca_out_w + (size_t)l * E_DIM * E_DIM, E_DIM, caout_t[l], E_DIM, E_DIM);
    cvtT(ffn_w1 + (size_t)l * E_DIM * F_DIM, F_DIM, ffn1_t[l], E_DIM, F_DIM);
    cvtT(ffn_w2 + (size_t)l * F_DIM * E_DIM, E_DIM, ffn2_t[l], F_DIM, E_DIM);
  }
  cvtT(be_w, E_DIM, be_t, E_DIM, E_DIM);
  cvtT(tp_w, E_DIM, tp_t, DT_DIM, E_DIM);
  cvtT(ct_qkv_w, 3 * E_DIM, ctqkv_t, E_DIM, 3 * E_DIM);
  cvtT(ct_out_w, E_DIM, ctout_t, E_DIM, E_DIM);
  cvtT(ce_w1, 2 * E_DIM, ce1_t, E_DIM, 2 * E_DIM);
  cvtT(ce_w2, E_DIM, ce2_t, 2 * E_DIM, E_DIM);
  cvtT(ip_w, E_DIM, ip_t, DI_DIM, E_DIM);
  cvtT(im_qkv_w, 3 * E_DIM, imqkv_t, E_DIM, 3 * E_DIM);
  cvtT(im_out_w, E_DIM, imout_t, E_DIM, E_DIM);
  cvtT(ie_w1, 2 * E_DIM, ie1_t, E_DIM, 2 * E_DIM);
  cvtT(ie_w2, E_DIM, ie2_t, 2 * E_DIM, E_DIM);
  cvtT(fin_w, V_SZ, fin_t, E_DIM, V_SZ);
  cvt(enc, enc_bf, (size_t)RS * E_DIM);

  // --- embedding ---
  embed_kernel<<<(R * E_DIM + 255) / 256, 256, 0, stream>>>(ids, item_emb, pos_emb, x);
  cvt(x, x_bf, (size_t)R * E_DIM);

  // --- decoder layers ---
  for (int l = 0; l < L_NUM; ++l) {
    // self-attention
    gemm(x_bf, E_DIM, saqkv_t[l], E_DIM, sa_qkv_b + (size_t)l * 3 * E_DIM,
         qkv, 3 * E_DIM, R, 3 * E_DIM, E_DIM, 0);
    attn_kernel<<<attn_grid, 256, 0, stream>>>(qkv, 3 * E_DIM, 0,
                                               qkv, 3 * E_DIM, E_DIM, 2 * E_DIM,
                                               nullptr, 1, ctx, T_LEN, T_LEN);
    cvt(ctx, ctx_bf, (size_t)R * E_DIM);
    gemm(ctx_bf, E_DIM, saout_t[l], E_DIM, sa_out_b + (size_t)l * E_DIM,
         proj, E_DIM, R, E_DIM, E_DIM, 0);
    add_ln_kernel<<<R, E_DIM, 0, stream>>>(x, proj, n1_s + l * E_DIM, n1_b + l * E_DIM, x);
    cvt(x, x_bf, (size_t)R * E_DIM);

    // cross-attention over encoder memory (Q slice = Wt rows 0..255, KV = rows 256..767)
    gemm(x_bf, E_DIM, caqkv_t[l], E_DIM, ca_qkv_b + (size_t)l * 3 * E_DIM,
         qkv, E_DIM, R, E_DIM, E_DIM, 0);
    gemm(enc_bf, E_DIM, caqkv_t[l] + (size_t)E_DIM * E_DIM, E_DIM,
         ca_qkv_b + (size_t)l * 3 * E_DIM + E_DIM, kvb, 2 * E_DIM, RS, 2 * E_DIM, E_DIM, 0);
    attn_kernel<<<attn_grid, 256, 0, stream>>>(qkv, E_DIM, 0,
                                               kvb, 2 * E_DIM, 0, E_DIM,
                                               pmask, 2, ctx, T_LEN, S_LEN);
    cvt(ctx, ctx_bf, (size_t)R * E_DIM);
    gemm(ctx_bf, E_DIM, caout_t[l], E_DIM, ca_out_b + (size_t)l * E_DIM,
         proj, E_DIM, R, E_DIM, E_DIM, 0);
    add_ln_kernel<<<R, E_DIM, 0, stream>>>(x, proj, n2_s + l * E_DIM, n2_b + l * E_DIM, x);
    cvt(x, x_bf, (size_t)R * E_DIM);

    // FFN
    gemm(x_bf, E_DIM, ffn1_t[l], E_DIM, ffn_b1 + (size_t)l * F_DIM,
         ffnh, F_DIM, R, F_DIM, E_DIM, 1);
    cvt(ffnh, ffnh_bf, (size_t)R * F_DIM);
    gemm(ffnh_bf, F_DIM, ffn2_t[l], F_DIM, ffn_b2 + (size_t)l * E_DIM,
         proj, E_DIM, R, E_DIM, F_DIM, 0);
    add_ln_kernel<<<R, E_DIM, 0, stream>>>(x, proj, n3_s + l * E_DIM, n3_b + l * E_DIM, x);
    cvt(x, x_bf, (size_t)R * E_DIM);
  }

  // --- MoE gate + behavior expert ---
  gate_kernel<<<R, GH_DIM, 0, stream>>>(x, g_w1, g_b1, g_w2, g_b2, gate);
  gemm(x_bf, E_DIM, be_t, E_DIM, be_b, ebeh, E_DIM, R, E_DIM, E_DIM, 0);

  // --- text expert ---
  gather_cvt_kernel<<<(R * DT_DIM + 255) / 256, 256, 0, stream>>>(text_mat, ids, DT_DIM, gath_bf, R);
  gemm(gath_bf, DT_DIM, tp_t, DT_DIM, tp_b, th, E_DIM, R, E_DIM, DT_DIM, 0);
  cvt(th, th_bf, (size_t)R * E_DIM);
  gemm(x_bf, E_DIM, ctqkv_t, E_DIM, ct_qkv_b, qkv, E_DIM, R, E_DIM, E_DIM, 0);
  gemm(th_bf, E_DIM, ctqkv_t + (size_t)E_DIM * E_DIM, E_DIM, ct_qkv_b + E_DIM,
       tkv, 2 * E_DIM, R, 2 * E_DIM, E_DIM, 0);
  attn_kernel<<<attn_grid, 256, 0, stream>>>(qkv, E_DIM, 0, tkv, 2 * E_DIM, 0, E_DIM,
                                             nullptr, 0, ctx, T_LEN, T_LEN);
  cvt(ctx, ctx_bf, (size_t)R * E_DIM);
  gemm(ctx_bf, E_DIM, ctout_t, E_DIM, ct_out_b, proj, E_DIM, R, E_DIM, E_DIM, 0);
  cvt(proj, proj_bf, (size_t)R * E_DIM);
  gemm(proj_bf, E_DIM, ce1_t, E_DIM, ce_b1, ehid, 2 * E_DIM, R, 2 * E_DIM, E_DIM, 1);
  cvt(ehid, ehid_bf, (size_t)R * 2 * E_DIM);
  gemm(ehid_bf, 2 * E_DIM, ce2_t, 2 * E_DIM, ce_b2, etxt, E_DIM, R, E_DIM, 2 * E_DIM, 0);

  // --- image expert ---
  gather_cvt_kernel<<<(R * DI_DIM + 255) / 256, 256, 0, stream>>>(img_mat, ids, DI_DIM, gath_bf, R);
  gemm(gath_bf, DI_DIM, ip_t, DI_DIM, ip_b, th, E_DIM, R, E_DIM, DI_DIM, 0);
  cvt(th, th_bf, (size_t)R * E_DIM);
  gemm(x_bf, E_DIM, imqkv_t, E_DIM, im_qkv_b, qkv, E_DIM, R, E_DIM, E_DIM, 0);
  gemm(th_bf, E_DIM, imqkv_t + (size_t)E_DIM * E_DIM, E_DIM, im_qkv_b + E_DIM,
       tkv, 2 * E_DIM, R, 2 * E_DIM, E_DIM, 0);
  attn_kernel<<<attn_grid, 256, 0, stream>>>(qkv, E_DIM, 0, tkv, 2 * E_DIM, 0, E_DIM,
                                             nullptr, 0, ctx, T_LEN, T_LEN);
  cvt(ctx, ctx_bf, (size_t)R * E_DIM);
  gemm(ctx_bf, E_DIM, imout_t, E_DIM, im_out_b, proj, E_DIM, R, E_DIM, E_DIM, 0);
  cvt(proj, proj_bf, (size_t)R * E_DIM);
  gemm(proj_bf, E_DIM, ie1_t, E_DIM, ie_b1, ehid, 2 * E_DIM, R, 2 * E_DIM, E_DIM, 1);
  cvt(ehid, ehid_bf, (size_t)R * 2 * E_DIM);
  gemm(ehid_bf, 2 * E_DIM, ie2_t, 2 * E_DIM, ie_b2, eimg, E_DIM, R, E_DIM, 2 * E_DIM, 0);

  // --- gated mix + full-vocab projection (memory-bound: 640 MB output) ---
  mix_kernel<<<(R * E_DIM + 255) / 256, 256, 0, stream>>>(gate, ebeh, etxt, eimg, hfin);
  cvt(hfin, hfin_bf, (size_t)R * E_DIM);
  gemm(hfin_bf, E_DIM, fin_t, E_DIM, fin_b, out, V_SZ, R, V_SZ, E_DIM, 0);
}
